// SparseDecoderBlock_69037304316039
// MI455X (gfx1250) — matrix operand order repaired and checked by reading, not compiled, not run on hardware
//
#include <hip/hip_runtime.h>

// Problem constants (match reference)
#define BB 2
#define SS 2048
#define DD 1024
#define HH 16
#define HD 64
#define KTOP 4
#define FF 4096
#define MM (BB*SS)   // 4096 rows

typedef __attribute__((ext_vector_type(16))) _Float16 v16h;
typedef __attribute__((ext_vector_type(8)))  float    v8f;

union FragH {
  unsigned int u[8];
  uint4 q[2];
  _Float16 h[16];
  v16h v;
};

__device__ __forceinline__ v8f wmma_f16(const FragH& a, const FragH& b, v8f c) {
  // D(f32 16x16) = A(16x32 f16) * B(32x16 f16) + C
  return __builtin_amdgcn_wmma_f32_16x16x32_f16(false, a.v, false, b.v, (short)0, c, false, false);
}

__device__ __forceinline__ v8f zero_v8() {
  v8f z;
#pragma unroll
  for (int i = 0; i < 8; i++) z[i] = 0.f;
  return z;
}

__device__ __forceinline__ float rowmax16(float v) {
  v = fmaxf(v, __shfl_xor(v, 1, 32));
  v = fmaxf(v, __shfl_xor(v, 2, 32));
  v = fmaxf(v, __shfl_xor(v, 4, 32));
  v = fmaxf(v, __shfl_xor(v, 8, 32));
  return v;
}
__device__ __forceinline__ float rowsum16(float v) {
  v += __shfl_xor(v, 1, 32);
  v += __shfl_xor(v, 2, 32);
  v += __shfl_xor(v, 4, 32);
  v += __shfl_xor(v, 8, 32);
  return v;
}

// ---------------- f32 -> f16 plain conversion ----------------
__global__ void cvt_f32_f16(const float* __restrict__ src, _Float16* __restrict__ dst, int n) {
  int i = blockIdx.x * blockDim.x + threadIdx.x;
  if (i < n) dst[i] = (_Float16)src[i];
}

// ---------------- f32 weight -> f16 B-fragment-swizzled layout ----------------
// Output block (kb, nb) = 32 lanes x 16 f16 contiguous; lane l holds
// column n = nb*16 + (l&15), k = kb*32 + (l>>4)*16 + e for e = 0..15.
// A B-fragment is then one aligned 32-byte load per lane.
__global__ void cvt_swz_b(const float* __restrict__ src, _Float16* __restrict__ dst,
                          int K, int N) {
  int t = blockIdx.x * blockDim.x + threadIdx.x;
  int total = (K >> 5) * (N >> 4) * 32;
  if (t >= total) return;
  int lane = t & 31;
  int blk = t >> 5;
  int nb = blk % (N >> 4);
  int kb = blk / (N >> 4);
  int n = nb * 16 + (lane & 15);
  int k0 = kb * 32 + (lane >> 4) * 16;
  FragH f;
#pragma unroll
  for (int e = 0; e < 16; e++) f.h[e] = (_Float16)src[(size_t)(k0 + e) * N + n];
  *(v16h*)(dst + (size_t)t * 16) = f.v;
}

// ---------------- RMSNorm (D=1024), writes f16 ----------------
__global__ __launch_bounds__(256)
void rmsnorm_f16(const float* __restrict__ x, const float* __restrict__ w,
                 _Float16* __restrict__ out) {
  __shared__ float red[8];
  const int row = blockIdx.x;
  const int tid = threadIdx.x;
  const int lane = tid & 31, wave = tid >> 5;
  const float* xr = x + (size_t)row * DD;
  float v[4];
  float ss = 0.f;
#pragma unroll
  for (int j = 0; j < 4; j++) { v[j] = xr[tid + j * 256]; ss += v[j] * v[j]; }
#pragma unroll
  for (int m = 16; m >= 1; m >>= 1) ss += __shfl_xor(ss, m, 32);
  if (lane == 0) red[wave] = ss;
  __syncthreads();
  float tot = 0.f;
#pragma unroll
  for (int wv = 0; wv < 8; wv++) tot += red[wv];
  float scale = rsqrtf(tot * (1.f / (float)DD) + 1e-6f);
  _Float16* orow = out + (size_t)row * DD;
#pragma unroll
  for (int j = 0; j < 4; j++)
    orow[tid + j * 256] = (_Float16)(v[j] * scale * w[tid + j * 256]);
}

// ---------------- LDS-free, double-buffered WMMA GEMM ----------------
// C[M,N] = A[M,K](f16 row-major) @ Bs(f16, fragment-swizzled).
// No divergent guards in the hot loop: B-block index is clamped, stores are
// guarded only in the epilogue (and only for the !FULL instantiation).
// MODE 0: Cf = acc     MODE 1: Cf = acc + addC     MODE 2: Ch = silu(gate)*acc
#define TM 128
#define TN 128
#define TK 32

template<int MODE, bool FULL>
__global__ __launch_bounds__(256)
void gemm_f16(const _Float16* __restrict__ A, const _Float16* __restrict__ Bs,
              int M, int N, int K,
              const float* __restrict__ addC, const float* __restrict__ gate,
              float* __restrict__ Cf, _Float16* __restrict__ Ch) {
  const int tid = threadIdx.x;
  const int lane = tid & 31, wave = tid >> 5;
  const int m0 = blockIdx.y * TM + (wave & 3) * 32;   // wave: 32 rows
  const int n0 = blockIdx.x * TN + (wave >> 2) * 64;  // wave: 64 cols
  const int NB = N >> 4;
  const int mm = lane & 15, kb = (lane >> 4) * 16;
  const int nb0 = n0 >> 4;

  v8f acc[2][4];
#pragma unroll
  for (int i = 0; i < 2; i++)
#pragma unroll
    for (int j = 0; j < 4; j++) acc[i][j] = zero_v8();

  FragH af[2][2], bf[2][4];

  auto loadf = [&](FragH (&a2)[2], FragH (&b4)[4], int kt) {
#pragma unroll
    for (int ia = 0; ia < 2; ia++) {
      const _Float16* pa = A + (size_t)(m0 + ia * 16 + mm) * K + kt;
      a2[ia].q[0] = *(const uint4*)(pa + kb);
      a2[ia].q[1] = *(const uint4*)(pa + 8 + kb);
    }
    const _Float16* pb = Bs + (size_t)(kt >> 5) * NB * 512 + (size_t)lane * 16;
#pragma unroll
    for (int jb = 0; jb < 4; jb++) {
      int nbe = FULL ? (nb0 + jb) : min(nb0 + jb, NB - 1);  // clamp: always in-bounds
      b4[jb].v = *(const v16h*)(pb + (size_t)nbe * 512);
    }
  };
  auto domath = [&](FragH (&a2)[2], FragH (&b4)[4]) {
#pragma unroll
    for (int jb = 0; jb < 4; jb++) {
      acc[0][jb] = wmma_f16(a2[0], b4[jb], acc[0][jb]);
      acc[1][jb] = wmma_f16(a2[1], b4[jb], acc[1][jb]);
    }
  };

  // K is always a multiple of 64, so the 2x-unrolled pipeline is exact.
  loadf(af[0], bf[0], 0);
  for (int kt = 0; kt < K; kt += 2 * TK) {
    loadf(af[1], bf[1], kt + TK);                  // prefetch next while computing
    __builtin_prefetch(A + (size_t)(m0 + mm) * K + kt + 2 * TK, 0, 1);
    domath(af[0], bf[0]);
    if (kt + 2 * TK < K) loadf(af[0], bf[0], kt + 2 * TK);
    domath(af[1], bf[1]);
  }

  // epilogue (C layout: lanes 0-15 rows 0-7, lanes 16-31 rows 8-15)
  const int nn = lane & 15;
  const int rbase = (lane >> 4) * 8;
#pragma unroll
  for (int ia = 0; ia < 2; ia++) {
#pragma unroll
    for (int jb = 0; jb < 4; jb++) {
      int gn = n0 + jb * 16 + nn;
      if (!FULL && gn >= N) continue;
#pragma unroll
      for (int r = 0; r < 8; r++) {
        int gm = m0 + ia * 16 + rbase + r;
        if (!FULL && gm >= M) continue;
        size_t idx = (size_t)gm * N + gn;
        float val = acc[ia][jb][r];
        if (MODE == 0) {
          Cf[idx] = val;
        } else if (MODE == 1) {
          Cf[idx] = val + addC[idx];
        } else {
          float g = gate[idx];
          float s = g / (1.f + __expf(-g));   // silu
          Ch[idx] = (_Float16)(s * val);
        }
      }
    }
  }
}

// ---------------- Router: argmax + running-count position (serial per (b,h)) ----------------
__global__ void router_kernel(const float* __restrict__ logits,
                              int* __restrict__ node, float* __restrict__ pos) {
  int t = threadIdx.x;
  if (t >= BB * HH) return;
  int b = t / HH, h = t % HH;
  int cnt[KTOP] = {0, 0, 0, 0};
  for (int s = 0; s < SS; s++) {
    size_t base = ((size_t)(b * SS + s) * HH + h) * KTOP;
    float best = logits[base];
    int idx = 0;
#pragma unroll
    for (int j = 1; j < KTOP; j++) {
      float lj = logits[base + j];
      if (lj > best) { best = lj; idx = j; }
    }
    cnt[idx]++;
    size_t o = (size_t)(b * SS + s) * HH + h;
    node[o] = idx;
    pos[o] = (float)(cnt[idx] - 1);
  }
}

// ---------------- RoPE on q,k + convert; V stored transposed [B][H][HD][S] ----------------
__global__ void rope_cvt_kernel(const float* __restrict__ q, const float* __restrict__ k,
                                const float* __restrict__ v, const float* __restrict__ pos,
                                _Float16* __restrict__ qh, _Float16* __restrict__ kh,
                                _Float16* __restrict__ vt) {
  int idx = blockIdx.x * blockDim.x + threadIdx.x;
  const int total = BB * SS * HH * (HD / 2);
  if (idx >= total) return;
  int i = idx & 31;
  int rest = idx >> 5;                 // (b*S+s)*H + h
  int h = rest % HH;
  int bs = rest / HH;
  int s = bs % SS;
  int b = bs / SS;
  float p = pos[rest];
  float inv = __expf(-(float)i * (9.210340371976184f / 32.f));  // 10000^(-i/32)
  float sn, cs;
  __sincosf(p * inv, &sn, &cs);
  size_t base = (size_t)rest * HD;
  float q1 = q[base + i], q2 = q[base + i + 32];
  qh[base + i]      = (_Float16)(q1 * cs - q2 * sn);
  qh[base + i + 32] = (_Float16)(q2 * cs + q1 * sn);
  float k1 = k[base + i], k2 = k[base + i + 32];
  kh[base + i]      = (_Float16)(k1 * cs - k2 * sn);
  kh[base + i + 32] = (_Float16)(k2 * cs + k1 * sn);
  size_t vbase = ((size_t)(b * HH + h) * HD + i) * SS + s;
  vt[vbase]                    = (_Float16)v[base + i];
  vt[vbase + (size_t)32 * SS]  = (_Float16)v[base + i + 32];
}

// ---------------- Flash attention: one wave per (b,h,16-query tile) ----------------
#define LP 40  // sP row stride in f16 (80B, 16B aligned)

__global__ __launch_bounds__(32)
void attn_kernel(const _Float16* __restrict__ qh, const _Float16* __restrict__ kh,
                 const _Float16* __restrict__ vt, const int* __restrict__ node,
                 float* __restrict__ ctx) {
  __shared__ _Float16 sP[16 * LP];
  const int lane = threadIdx.x;
  const int qt = blockIdx.x;
  const int bh = blockIdx.y;
  const int b = bh / HH, h = bh % HH;
  const int nn = lane & 15;
  const int half = lane >> 4;
  const int rbase = half * 8;
  const int q0 = qt * 16;
  const size_t rowstride = (size_t)HH * HD;

  int nq[8];
#pragma unroll
  for (int r = 0; r < 8; r++)
    nq[r] = node[(size_t)(b * SS + q0 + rbase + r) * HH + h];

  // Q as A fragments (d 0..31 and 32..63): two 16B runs each
  FragH qa[2];
  {
    int kb = half * 16;
    const _Float16* pq = qh + ((size_t)(b * SS + q0 + nn) * HH + h) * HD;
#pragma unroll
    for (int c = 0; c < 2; c++) {
      qa[c].q[0] = *(const uint4*)(pq + c * 32 + kb);
      qa[c].q[1] = *(const uint4*)(pq + c * 32 + 8 + kb);
    }
  }

  float mrow[8], lrow[8];
  v8f o[4];
#pragma unroll
  for (int r = 0; r < 8; r++) { mrow[r] = -1e30f; lrow[r] = 0.f; }
#pragma unroll
  for (int c = 0; c < 4; c++) o[c] = zero_v8();

  const size_t vtrow = ((size_t)(b * HH + h) * HD) * SS;  // vt base for this (b,h)
  const int dh = half * 16;
  const int nblocks = (qt >> 1) + 1;  // 32-key blocks, causal bound

  for (int jb = 0; jb < nblocks; jb++) {
    const int j0 = jb * 32;
    // ---- scores: S = Q @ K^T, two 16-key halves; K frags are 32B contiguous ----
    v8f s0 = zero_v8(), s1 = zero_v8();
    {
      const _Float16* pk0 = kh + ((size_t)(b * SS + j0 + nn) * HH + h) * HD;
      const _Float16* pk1 = pk0 + 16 * rowstride;
#pragma unroll
      for (int c = 0; c < 2; c++) {
        FragH bf0, bf1;
        bf0.v = *(const v16h*)(pk0 + c * 32 + dh);
        bf1.v = *(const v16h*)(pk1 + c * 32 + dh);
        s0 = wmma_f16(qa[c], bf0, s0);
        s1 = wmma_f16(qa[c], bf1, s1);
      }
    }
    // ---- mask + online softmax ----
    int nk0 = node[(size_t)(b * SS + j0 + nn) * HH + h];
    int nk1 = node[(size_t)(b * SS + j0 + 16 + nn) * HH + h];
    int key0 = j0 + nn, key1 = j0 + 16 + nn;
    float pr0[8], pr1[8];
#pragma unroll
    for (int r = 0; r < 8; r++) {
      int qrow = q0 + rbase + r;
      bool a0 = (key0 <= qrow) && (nk0 == nq[r]);
      bool a1 = (key1 <= qrow) && (nk1 == nq[r]);
      float v0 = a0 ? s0[r] * 0.125f : -1e30f;   // 1/sqrt(64)
      float v1 = a1 ? s1[r] * 0.125f : -1e30f;
      float t = rowmax16(fmaxf(v0, v1));
      float mnew = fmaxf(mrow[r], t);
      float scale = __expf(mrow[r] - mnew);
      mrow[r] = mnew;
      float p0 = a0 ? __expf(v0 - mnew) : 0.f;
      float p1 = a1 ? __expf(v1 - mnew) : 0.f;
      lrow[r] = lrow[r] * scale + rowsum16(p0 + p1);
#pragma unroll
      for (int c = 0; c < 4; c++) o[c][r] = o[c][r] * scale;
      pr0[r] = p0; pr1[r] = p1;
    }
    // ---- P (C layout) -> LDS -> A layout ----
#pragma unroll
    for (int r = 0; r < 8; r++) {
      sP[(rbase + r) * LP + nn]      = (_Float16)pr0[r];
      sP[(rbase + r) * LP + 16 + nn] = (_Float16)pr1[r];
    }
    asm volatile("s_wait_dscnt 0" ::: "memory");
    FragH pf;
    {
      int kb = half * 16;
      const _Float16* pp = sP + nn * LP;
      pf.q[0] = *(const uint4*)(pp + kb);
      pf.q[1] = *(const uint4*)(pp + 8 + kb);
    }
    // ---- O += P @ V ; V frags are 32B contiguous in transposed V ----
#pragma unroll
    for (int c = 0; c < 4; c++) {
      FragH vf;
      vf.v = *(const v16h*)(vt + vtrow + (size_t)(c * 16 + nn) * SS + j0 + half * 16);
      o[c] = wmma_f16(pf, vf, o[c]);
    }
  }

  // ---- normalize + store ctx (B,S,H*HD) ----
#pragma unroll
  for (int r = 0; r < 8; r++) {
    float invl = 1.f / lrow[r];
    size_t cb = ((size_t)(b * SS + q0 + rbase + r) * HH + h) * HD;
#pragma unroll
    for (int c = 0; c < 4; c++)
      ctx[cb + c * 16 + nn] = o[c][r] * invl;
  }
}

// ------------------------------------------------------------------
extern "C" void kernel_launch(void* const* d_in, const int* in_sizes, int n_in,
                              void* d_out, int out_size, void* d_ws, size_t ws_size,
                              hipStream_t stream) {
  const float* x   = (const float*)d_in[0];
  const float* anw = (const float*)d_in[1];
  const float* fnw = (const float*)d_in[2];
  const float* Wq  = (const float*)d_in[3];
  const float* Wk  = (const float*)d_in[4];
  const float* Wv  = (const float*)d_in[5];
  const float* Wo  = (const float*)d_in[6];
  const float* Wr  = (const float*)d_in[7];
  const float* w1  = (const float*)d_in[8];
  const float* w2  = (const float*)d_in[9];
  const float* w3  = (const float*)d_in[10];
  float* out = (float*)d_out;

  char* ws = (char*)d_ws;
  size_t off = 0;
  auto alloc = [&](size_t bytes) -> char* {
    char* p = ws + off;
    off += (bytes + 255) & ~(size_t)255;
    return p;
  };

  const size_t NQ = (size_t)MM * HH * HD;  // 4M elems
  _Float16* Wq_h = (_Float16*)alloc((size_t)DD * HH * HD * 2);
  _Float16* Wk_h = (_Float16*)alloc((size_t)DD * HH * HD * 2);
  _Float16* Wv_h = (_Float16*)alloc((size_t)DD * HH * HD * 2);
  _Float16* Wo_h = (_Float16*)alloc((size_t)HH * HD * DD * 2);
  _Float16* Wr_h = (_Float16*)alloc((size_t)DD * HH * KTOP * 2);
  _Float16* w1_h = (_Float16*)alloc((size_t)DD * FF * 2);
  _Float16* w2_h = (_Float16*)alloc((size_t)DD * FF * 2);
  _Float16* w3_h = (_Float16*)alloc((size_t)FF * DD * 2);
  _Float16* hb   = (_Float16*)alloc((size_t)MM * DD * 2);
  float* logits  = (float*)alloc((size_t)MM * HH * KTOP * 4);
  int*   node    = (int*)alloc((size_t)MM * HH * 4);
  float* pos     = (float*)alloc((size_t)MM * HH * 4);
  float* x1      = (float*)alloc((size_t)MM * DD * 4);
  _Float16* h2b  = (_Float16*)alloc((size_t)MM * DD * 2);
  // BIGA (64MB f32): q|k|v|ctx, later aliased by g1 (M x F)
  float* biga = (float*)alloc((size_t)MM * FF * 4);
  float* qf = biga, *kf = biga + NQ, *vf = biga + 2 * NQ, *ctxf = biga + 3 * NQ;
  float* g1 = biga;
  // BIGB (32MB f16): qh|kh|vt|ctxh, later aliased by ub (M x F)
  _Float16* bigb = (_Float16*)alloc((size_t)MM * FF * 2);
  _Float16* qhh = bigb, *khh = bigb + NQ, *vth = bigb + 2 * NQ, *ctxh = bigb + 3 * NQ;
  _Float16* ub = bigb;

  auto cgrid = [](size_t n) { return dim3((unsigned)((n + 255) / 256)); };

  // --- weights to f16, B-fragment swizzled (K,N per first use) ---
  cvt_swz_b<<<cgrid((size_t)DD * HH * HD / 16), 256, 0, stream>>>(Wq, Wq_h, DD, HH * HD);
  cvt_swz_b<<<cgrid((size_t)DD * HH * HD / 16), 256, 0, stream>>>(Wk, Wk_h, DD, HH * HD);
  cvt_swz_b<<<cgrid((size_t)DD * HH * HD / 16), 256, 0, stream>>>(Wv, Wv_h, DD, HH * HD);
  cvt_swz_b<<<cgrid((size_t)HH * HD * DD / 16), 256, 0, stream>>>(Wo, Wo_h, HH * HD, DD);
  cvt_swz_b<<<cgrid((size_t)DD * HH * KTOP / 16), 256, 0, stream>>>(Wr, Wr_h, DD, HH * KTOP);
  cvt_swz_b<<<cgrid((size_t)DD * FF / 16), 256, 0, stream>>>(w1, w1_h, DD, FF);
  cvt_swz_b<<<cgrid((size_t)DD * FF / 16), 256, 0, stream>>>(w2, w2_h, DD, FF);
  cvt_swz_b<<<cgrid((size_t)FF * DD / 16), 256, 0, stream>>>(w3, w3_h, FF, DD);

  // --- attention half ---
  rmsnorm_f16<<<MM, 256, 0, stream>>>(x, anw, hb);
  gemm_f16<0, false><<<dim3((HH * KTOP + TN - 1) / TN, MM / TM), 256, 0, stream>>>(
      hb, Wr_h, MM, HH * KTOP, DD, nullptr, nullptr, logits, nullptr);
  gemm_f16<0, true><<<dim3((HH * HD) / TN, MM / TM), 256, 0, stream>>>(
      hb, Wq_h, MM, HH * HD, DD, nullptr, nullptr, qf, nullptr);
  gemm_f16<0, true><<<dim3((HH * HD) / TN, MM / TM), 256, 0, stream>>>(
      hb, Wk_h, MM, HH * HD, DD, nullptr, nullptr, kf, nullptr);
  gemm_f16<0, true><<<dim3((HH * HD) / TN, MM / TM), 256, 0, stream>>>(
      hb, Wv_h, MM, HH * HD, DD, nullptr, nullptr, vf, nullptr);
  router_kernel<<<1, 32, 0, stream>>>(logits, node, pos);
  rope_cvt_kernel<<<cgrid((size_t)MM * HH * (HD / 2)), 256, 0, stream>>>(
      qf, kf, vf, pos, qhh, khh, vth);
  attn_kernel<<<dim3(SS / 16, BB * HH), 32, 0, stream>>>(qhh, khh, vth, node, ctxf);
  cvt_f32_f16<<<cgrid(NQ), 256, 0, stream>>>(ctxf, ctxh, (int)NQ);
  gemm_f16<1, true><<<dim3(DD / TN, MM / TM), 256, 0, stream>>>(
      ctxh, Wo_h, MM, DD, HH * HD, x, nullptr, x1, nullptr);   // x1 = x + ctx@Wo

  // --- FFN half ---
  rmsnorm_f16<<<MM, 256, 0, stream>>>(x1, fnw, h2b);
  gemm_f16<0, true><<<dim3(FF / TN, MM / TM), 256, 0, stream>>>(
      h2b, w1_h, MM, FF, DD, nullptr, nullptr, g1, nullptr);   // g1 = h2@w1
  gemm_f16<2, true><<<dim3(FF / TN, MM / TM), 256, 0, stream>>>(
      h2b, w2_h, MM, FF, DD, nullptr, g1, nullptr, ub);        // ub = silu(g1)*(h2@w2)
  gemm_f16<1, true><<<dim3(DD / TN, MM / TM), 256, 0, stream>>>(
      ub, w3_h, MM, DD, FF, x1, nullptr, out, nullptr);        // out = x1 + ub@w3

  (void)in_sizes; (void)n_in; (void)out_size; (void)ws_size;
}
